// GraphormerAttentionHead_24464133718314
// MI455X (gfx1250) — compile-verified, matrix operand order checked by reading, not run on hardware
//
#include <hip/hip_runtime.h>
#include <hip/hip_bf16.h>

typedef __attribute__((ext_vector_type(16))) _Float16 v16h;
typedef __attribute__((ext_vector_type(8)))  _Float16 v8h;
typedef __attribute__((ext_vector_type(4)))  _Float16 v4h;
typedef __attribute__((ext_vector_type(8)))  float    v8f;

#define N_NODES 8192
#define DDIM    512
#define NEGVAL  (-1000000.0f)
#define ADJ_W   (N_NODES / 32)
#define PANEL   256              // keys per attention iteration

// ISA 7.12.2 wave32 16-bit fragment: lane's 16 halves are two contiguous runs
// of 8 elements: k = g*8 + [0,8) and k = g*8 + 16 + [0,8), g = lane/16.
// -> two 16-byte loads per fragment.
__device__ __forceinline__ v16h load_frag(const _Float16* p) {
    v8h lo = *(const v8h*)(p);
    v8h hi = *(const v8h*)(p + 16);
    return __builtin_shufflevector(lo, hi, 0, 1, 2, 3, 4, 5, 6, 7,
                                           8, 9, 10, 11, 12, 13, 14, 15);
}

__device__ __forceinline__ v8f wmma_f16(v16h a, v16h b, v8f c) {
    return __builtin_amdgcn_wmma_f32_16x16x32_f16(
               false, a, false, b, (short)0, c, false, false);
}

// ---------------------------------------------------------------- adjacency
__global__ void zero_adj_kernel(unsigned* __restrict__ adj, int nwords) {
    for (int i = blockIdx.x * blockDim.x + threadIdx.x; i < nwords;
         i += gridDim.x * blockDim.x)
        adj[i] = 0u;
}

__global__ void scatter_adj_kernel(const int* __restrict__ ei,
                                   unsigned* __restrict__ adj, int E) {
    int e = blockIdx.x * blockDim.x + threadIdx.x;
    if (e < E) {
        int r = ei[e]     & (N_NODES - 1);
        int c = ei[E + e] & (N_NODES - 1);
        atomicOr(&adj[r * ADJ_W + (c >> 5)], 1u << (c & 31));
    }
}

// ---------------------------------------------------------------- prepass
__global__ void cvt_x_kernel(const float* __restrict__ x,
                             _Float16* __restrict__ xh, int n) {
    int i = (blockIdx.x * blockDim.x + threadIdx.x) * 4;
    if (i < n) {
        float4 f = *(const float4*)(x + i);
        v4h o = { (_Float16)f.x, (_Float16)f.y, (_Float16)f.z, (_Float16)f.w };
        *(v4h*)(xh + i) = o;
    }
}

__global__ void cvt_w_kernel(const float* __restrict__ Wq,
                             const float* __restrict__ Wk,
                             const float* __restrict__ Wv,
                             _Float16* __restrict__ Wt) {
    const float* W = (blockIdx.y == 0) ? Wq : (blockIdx.y == 1) ? Wk : Wv;
    _Float16*   Wo = Wt + (size_t)blockIdx.y * DDIM * DDIM;
    int t = blockIdx.x * blockDim.x + threadIdx.x;   // t = n*512 + k
    if (t < DDIM * DDIM) {
        int n = t >> 9, k = t & (DDIM - 1);
        Wo[t] = (_Float16)W[k * DDIM + n];
    }
}

// ---------------------------------------------------------------- QKV GEMM
// One 32x32 tile per wave (2x2 register blocking): 8 b128 loads -> 4 WMMAs.
__global__ __launch_bounds__(256) void qkv_gemm_kernel(
    const _Float16* __restrict__ xh, const _Float16* __restrict__ Wt,
    const float* __restrict__ bq, const float* __restrict__ bk,
    const float* __restrict__ bv,
    _Float16* __restrict__ Q, _Float16* __restrict__ K,
    _Float16* __restrict__ Vt)
{
    const int mode = blockIdx.z;
    const _Float16* __restrict__ W = Wt + (size_t)mode * DDIM * DDIM;
    const float* __restrict__ b = (mode == 0) ? bq : (mode == 1) ? bk : bv;

    const int lane = threadIdx.x & 31;
    const int wave = threadIdx.x >> 5;
    const int idx  = lane & 15;
    const int g    = (lane >> 4) & 1;

    const int tile = blockIdx.x * 8 + wave;   // 4096 tiles = 256 x 16
    const int m0   = (tile >> 4) << 5;
    const int n0   = (tile & 15) << 5;

    const _Float16* ap0 = xh + (size_t)(m0 + idx) * DDIM + (g << 3);
    const _Float16* ap1 = ap0 + (size_t)16 * DDIM;
    const _Float16* bp0 = W  + (size_t)(n0 + idx) * DDIM + (g << 3);
    const _Float16* bp1 = bp0 + (size_t)16 * DDIM;

    v8f a00 = {}, a01 = {}, a10 = {}, a11 = {};
    #pragma unroll 2
    for (int kk = 0; kk < DDIM; kk += 32) {
        v16h fa0 = load_frag(ap0 + kk);
        v16h fa1 = load_frag(ap1 + kk);
        v16h fb0 = load_frag(bp0 + kk);
        v16h fb1 = load_frag(bp1 + kk);
        a00 = wmma_f16(fa0, fb0, a00);
        a01 = wmma_f16(fa0, fb1, a01);
        a10 = wmma_f16(fa1, fb0, a10);
        a11 = wmma_f16(fa1, fb1, a11);
    }

    const float bias0  = b[n0 + idx];
    const float bias1  = b[n0 + 16 + idx];
    const float qscale = 0.044194173824159216f;  // 1/sqrt(512)

    if (mode == 2) {
        v8h p00, p01, p10, p11;
        #pragma unroll
        for (int r = 0; r < 8; ++r) {
            p00[r] = (_Float16)(a00[r] + bias0);
            p10[r] = (_Float16)(a10[r] + bias0);
            p01[r] = (_Float16)(a01[r] + bias1);
            p11[r] = (_Float16)(a11[r] + bias1);
        }
        _Float16* v0 = Vt + (size_t)(n0 + idx) * N_NODES + m0 + (g << 3);
        _Float16* v1 = Vt + (size_t)(n0 + 16 + idx) * N_NODES + m0 + (g << 3);
        *(v8h*)(v0)      = p00;
        *(v8h*)(v0 + 16) = p10;
        *(v8h*)(v1)      = p01;
        *(v8h*)(v1 + 16) = p11;
    } else {
        _Float16* dst = (mode == 0) ? Q : K;
        const float sc = (mode == 0) ? qscale : 1.0f;
        #pragma unroll
        for (int r = 0; r < 8; ++r) {
            int m = r + (g << 3);
            dst[(size_t)(m0 + m) * DDIM + n0 + idx]      = (_Float16)((a00[r] + bias0) * sc);
            dst[(size_t)(m0 + m) * DDIM + n0 + 16 + idx] = (_Float16)((a01[r] + bias1) * sc);
            dst[(size_t)(m0 + 16 + m) * DDIM + n0 + idx]      = (_Float16)((a10[r] + bias0) * sc);
            dst[(size_t)(m0 + 16 + m) * DDIM + n0 + 16 + idx] = (_Float16)((a11[r] + bias1) * sc);
        }
    }
}

// ---------------------------------------------------------------- attention
// One block per 16 query rows. 256-key panels: each of the 8 waves owns 32
// keys (shared A-frag, 2 S tiles), block-wide online softmax via LDS, then
// each wave accumulates its 64 output columns with P @ V.
__global__ __launch_bounds__(256) void attn_kernel(
    const _Float16* __restrict__ Q, const _Float16* __restrict__ K,
    const _Float16* __restrict__ Vt, const unsigned* __restrict__ adj,
    float* __restrict__ out)
{
    __shared__ __align__(16) _Float16 Qs[16 * DDIM];    // 16 KB
    __shared__ __align__(16) float    Ss[16 * PANEL];   // 16 KB
    __shared__ __align__(16) _Float16 Ps[16 * PANEL];   //  8 KB
    __shared__ float red[16 * 16];
    __shared__ float m_run[16], l_run[16], r_scale[16];

    const int tid  = threadIdx.x;
    const int lane = tid & 31;
    const int wave = tid >> 5;
    const int idx  = lane & 15;
    const int g    = (lane >> 4) & 1;
    const int q0   = blockIdx.x << 4;

    // stage Q tile into LDS via the gfx1250 async global->LDS DMA path
    for (int i = tid * 8; i < 16 * DDIM; i += 256 * 8) {
        unsigned lds = (unsigned)(uintptr_t)(Qs + i);
        const _Float16* gp =
            Q + (size_t)(q0 + (i >> 9)) * DDIM + (i & (DDIM - 1));
        asm volatile("global_load_async_to_lds_b128 %0, %1, off"
                     :: "v"(lds), "v"(gp) : "memory");
    }
    asm volatile("s_wait_asynccnt 0x0" ::: "memory");
    if (tid < 16) { m_run[tid] = -3.0e38f; l_run[tid] = 0.0f; }
    __syncthreads();

    v8f o0 = {}, o1 = {}, o2 = {}, o3 = {};
    const int c0 = wave << 6;   // this wave's 64 output columns

    for (int kb = 0; kb < N_NODES; kb += PANEL) {
        // prefetch next panel (K rows + Vt columns); K/Vt are L2-resident
        if (kb + PANEL < N_NODES) {
            __builtin_prefetch(
                K + (size_t)(kb + PANEL + tid) * DDIM + ((tid & 1) << 8), 0, 1);
            __builtin_prefetch(
                Vt + (size_t)(tid * 2) * N_NODES + kb + PANEL, 0, 1);
        }

        // ---- S tiles: this wave owns keys [key0, key0+32)
        const int key0 = kb + (wave << 5);
        const _Float16* aq  = Qs + idx * DDIM + (g << 3);
        const _Float16* bk0 = K + (size_t)(key0 + idx) * DDIM + (g << 3);
        const _Float16* bk1 = bk0 + (size_t)16 * DDIM;
        v8f s0 = {}, s1 = {};
        #pragma unroll 4
        for (int dk = 0; dk < DDIM; dk += 32) {
            v16h a  = load_frag(aq + dk);
            v16h f0 = load_frag(bk0 + dk);
            v16h f1 = load_frag(bk1 + dk);
            s0 = wmma_f16(a, f0, s0);
            s1 = wmma_f16(a, f1, s1);
        }
        #pragma unroll
        for (int r = 0; r < 8; ++r) {
            int m    = r + (g << 3);
            int keyA = key0 + idx;
            int keyB = keyA + 16;
            const unsigned* arow = adj + (size_t)(q0 + m) * ADJ_W;
            unsigned bitA = (arow[keyA >> 5] >> (keyA & 31)) & 1u;
            unsigned bitB = (arow[keyB >> 5] >> (keyB & 31)) & 1u;
            Ss[m * PANEL + (wave << 5) + idx]      = bitA ? s0[r] : NEGVAL;
            Ss[m * PANEL + (wave << 5) + 16 + idx] = bitB ? s1[r] : NEGVAL;
        }
        __syncthreads();

        // ---- online softmax over the 256-key panel (16 cols per thread)
        const int row = tid >> 4, ch = tid & 15;
        {
            float lm = -3.0e38f;
            #pragma unroll
            for (int j = 0; j < 16; ++j)
                lm = fmaxf(lm, Ss[row * PANEL + ch * 16 + j]);
            red[row * 16 + ch] = lm;
        }
        __syncthreads();
        if (tid < 16) {
            float mx = -3.0e38f;
            #pragma unroll
            for (int j = 0; j < 16; ++j) mx = fmaxf(mx, red[tid * 16 + j]);
            float mnew   = fmaxf(m_run[tid], mx);
            r_scale[tid] = __expf(m_run[tid] - mnew);
            m_run[tid]   = mnew;
        }
        __syncthreads();
        {
            float ls   = 0.0f;
            float mrow = m_run[row];
            #pragma unroll
            for (int j = 0; j < 16; ++j) {
                float p = __expf(Ss[row * PANEL + ch * 16 + j] - mrow);
                Ps[row * PANEL + ch * 16 + j] = (_Float16)p;
                ls += p;
            }
            red[row * 16 + ch] = ls;
        }
        __syncthreads();
        if (tid < 16) {
            float sum = 0.0f;
            #pragma unroll
            for (int j = 0; j < 16; ++j) sum += red[tid * 16 + j];
            l_run[tid] = l_run[tid] * r_scale[tid] + sum;
        }
        __syncthreads();

        // ---- rescale O and accumulate P @ V (cols [c0, c0+64))
        #pragma unroll
        for (int r = 0; r < 8; ++r) {
            float f = r_scale[r + (g << 3)];
            o0[r] *= f; o1[r] *= f; o2[r] *= f; o3[r] *= f;
        }
        #pragma unroll
        for (int kc = 0; kc < PANEL; kc += 32) {
            v16h a = load_frag(Ps + idx * PANEL + kc + (g << 3));
            const _Float16* vb =
                Vt + (size_t)(c0 + idx) * N_NODES + kb + kc + (g << 3);
            v16h b0 = load_frag(vb);
            v16h b1 = load_frag(vb + (size_t)16 * N_NODES);
            v16h b2 = load_frag(vb + (size_t)32 * N_NODES);
            v16h b3 = load_frag(vb + (size_t)48 * N_NODES);
            o0 = wmma_f16(a, b0, o0);
            o1 = wmma_f16(a, b1, o1);
            o2 = wmma_f16(a, b2, o2);
            o3 = wmma_f16(a, b3, o3);
        }
        __syncthreads();
    }

    // ---- normalize and write out (f32)
    #pragma unroll
    for (int r = 0; r < 8; ++r) {
        int    m    = r + (g << 3);
        float  inv  = 1.0f / l_run[m];
        size_t base = (size_t)(q0 + m) * DDIM + c0 + idx;
        out[base +  0] = o0[r] * inv;
        out[base + 16] = o1[r] * inv;
        out[base + 32] = o2[r] * inv;
        out[base + 48] = o3[r] * inv;
    }
}

// ---------------------------------------------------------------- launcher
extern "C" void kernel_launch(void* const* d_in, const int* in_sizes, int n_in,
                              void* d_out, int out_size, void* d_ws, size_t ws_size,
                              hipStream_t stream) {
    const float* x  = (const float*)d_in[0];
    const int*   ei = (const int*)d_in[1];
    const float* Wq = (const float*)d_in[2];
    const float* bq = (const float*)d_in[3];
    const float* Wk = (const float*)d_in[4];
    const float* bk = (const float*)d_in[5];
    const float* Wv = (const float*)d_in[6];
    const float* bv = (const float*)d_in[7];
    float* out = (float*)d_out;
    const int E = in_sizes[1] / 2;

    // Workspace layout:
    //   [0,  8M): adjacency bitmask u32[8192*256]
    //   [8M, 16M): xh f16 [8192][512]
    //   [16M,24M): Q  f16 [8192][512] (pre-scaled by 1/sqrt(512))
    //   [24M,32M): K  f16 [8192][512] (natural layout)
    //   [32M,40M): Vt f16 [512][8192] (transposed)
    //   [40M,41.5M): Wt f16 3x[512][512] (transposed weights)
    char* ws = (char*)d_ws;
    unsigned* adj = (unsigned*)ws;
    _Float16* xh  = (_Float16*)(ws + ((size_t)8  << 20));
    _Float16* Q   = (_Float16*)(ws + ((size_t)16 << 20));
    _Float16* K   = (_Float16*)(ws + ((size_t)24 << 20));
    _Float16* Vt  = (_Float16*)(ws + ((size_t)32 << 20));
    _Float16* Wt  = (_Float16*)(ws + ((size_t)40 << 20));

    const int nwords = N_NODES * ADJ_W;
    zero_adj_kernel<<<2048, 256, 0, stream>>>(adj, nwords);
    scatter_adj_kernel<<<(E + 255) / 256, 256, 0, stream>>>(ei, adj, E);

    const int nx = N_NODES * DDIM;
    cvt_x_kernel<<<(nx / 4 + 255) / 256, 256, 0, stream>>>(x, xh, nx);
    cvt_w_kernel<<<dim3((DDIM * DDIM + 255) / 256, 3), 256, 0, stream>>>(
        Wq, Wk, Wv, Wt);

    qkv_gemm_kernel<<<dim3(512, 1, 3), 256, 0, stream>>>(
        xh, Wt, bq, bk, bv, Q, K, Vt);
    attn_kernel<<<512, 256, 0, stream>>>(Q, K, Vt, adj, out);
}
